// Attention_94489280516
// MI455X (gfx1250) — matrix operand order repaired and checked by reading, not compiled, not run on hardware
//
#include <hip/hip_runtime.h>
#include <hip/hip_bf16.h>

typedef __bf16 bf16;
typedef __attribute__((ext_vector_type(16))) __bf16 v16bf;
typedef __attribute__((ext_vector_type(8)))  __bf16 v8bf;
typedef __attribute__((ext_vector_type(8)))  float  v8f;

#define CHN   256
#define LSEQ  8192
#define INV_L (1.0f / 8192.0f)

// ---- workspace layout (bytes) ----
static const size_t OFF_XB   = 0;                                   // x  bf16 [8192,256]   4 MB
static const size_t OFF_WQB  = OFF_XB  + (size_t)LSEQ * CHN * 2;    // Wq bf16 [256,256]  128 KB
static const size_t OFF_WKB  = OFF_WQB + (size_t)CHN * CHN * 2;
static const size_t OFF_WVB  = OFF_WKB + (size_t)CHN * CHN * 2;
static const size_t OFF_QB   = OFF_WVB + (size_t)CHN * CHN * 2;     // Q bf16 [8192,256]
static const size_t OFF_KB   = OFF_QB  + (size_t)LSEQ * CHN * 2;    // K bf16 [8192,256]
static const size_t OFF_VT   = OFF_KB  + (size_t)LSEQ * CHN * 2;    // V^T bf16 [256,8192]
static const size_t OFF_PART = OFF_VT  + (size_t)LSEQ * CHN * 2;    // 4096 (m,S) pairs
static const size_t OFF_SCAL = OFF_PART + (size_t)4096 * 2 * 4;     // global m, S

// ---------------- WMMA helpers (CDNA5 layouts, CDNA5 ISA §7.12.2) ----------------

__device__ __forceinline__ v8f wmma_bf16f32(v16bf a, v16bf b, v8f c) {
    // D = A(16x32 bf16) * B(32x16 bf16) + C(16x16 f32)
    return __builtin_amdgcn_wmma_f32_16x16x32_bf16(
        /*neg_a=*/false, a, /*neg_b=*/false, b,
        /*c_mod=*/(short)0, c, /*reuse_a=*/false, /*reuse_b=*/false);
}

// A fragment: lane l holds row M = l%16; lanes 0-15 carry K = {k0..k0+7, k0+16..k0+23},
// lanes 16-31 carry K = {k0+8..k0+15, k0+24..k0+31}.  Two 16-byte contiguous loads.
__device__ __forceinline__ v16bf load_a_frag(const bf16* M, int ld, int row0, int k0, int lane) {
    int r = lane & 15, h = lane >> 4;
    const bf16* p = M + (size_t)(row0 + r) * ld + k0 + h * 8;
    v8bf lo = *(const v8bf*)p;
    v8bf hi = *(const v8bf*)(p + 16);
    return __builtin_shufflevector(lo, hi, 0,1,2,3,4,5,6,7,8,9,10,11,12,13,14,15);
}

// B fragment: lane l holds col N = l%16; lanes 0-15 carry K = {k0..k0+7, k0+16..k0+23},
// lanes 16-31 carry K = {k0+8..k0+15, k0+24..k0+31}.  Bsrc[n][k] row-major (i.e. B already "column rows").
__device__ __forceinline__ v16bf load_b_frag(const bf16* Bsrc, size_t ld, int n0, int k0, int lane) {
    int c = lane & 15, h = lane >> 4;
    const bf16* p = Bsrc + (size_t)(n0 + c) * ld + k0 + h * 8;
    v8bf lo = *(const v8bf*)p;
    v8bf hi = *(const v8bf*)(p + 16);
    return __builtin_shufflevector(lo, hi, 0,1,2,3,4,5,6,7,8,9,10,11,12,13,14,15);
}

// ---------------- Kernel 0: f32 -> bf16 conversion of x, Wq, Wk, Wv ----------------

__global__ void k_prep(const float* __restrict__ x,  const float* __restrict__ wq,
                       const float* __restrict__ wk, const float* __restrict__ wv,
                       bf16* xb, bf16* wqb, bf16* wkb, bf16* wvb) {
    const int NX = LSEQ * CHN, NW = CHN * CHN;
    for (int i = blockIdx.x * blockDim.x + threadIdx.x; i < NX + 3 * NW;
         i += gridDim.x * blockDim.x) {
        if (i < NX)                xb[i]              = (bf16)x[i];
        else if (i < NX + NW)      wqb[i - NX]        = (bf16)wq[i - NX];
        else if (i < NX + 2 * NW)  wkb[i - NX - NW]   = (bf16)wk[i - NX - NW];
        else                       wvb[i - NX - 2*NW] = (bf16)wv[i - NX - 2*NW];
    }
}

// ---------------- Kernel 1: Q/K/V projection, y = x @ W^T + b (WMMA) ----------------
// blockIdx.y selects matrix (0=Q row-major, 1=K row-major, 2=V stored transposed).

__global__ void k_proj(const bf16* __restrict__ xb,
                       const bf16* __restrict__ wqb, const bf16* __restrict__ wkb,
                       const bf16* __restrict__ wvb,
                       const float* __restrict__ bq, const float* __restrict__ bk,
                       const float* __restrict__ bv,
                       bf16* qb, bf16* kb, bf16* vT) {
    int lane = threadIdx.x & 31, wave = threadIdx.x >> 5;
    int mat = blockIdx.y;
    const bf16*  W    = (mat == 0) ? wqb : (mat == 1) ? wkb : wvb;
    const float* bias = (mat == 0) ? bq  : (mat == 1) ? bk  : bv;

    int tile = blockIdx.x * 8 + wave;       // 8192 tiles of 16x16
    int row0 = (tile >> 4) * 16;            // token rows
    int col0 = (tile & 15) * 16;            // output channels

    v8f d = {};
    for (int k = 0; k < 8; ++k) {           // K = 256 = 8 x 32
        v16bf a = load_a_frag(xb, CHN, row0, k * 32, lane);
        v16bf b = load_b_frag(W,  CHN, col0, k * 32, lane);   // B[k,c] = W[c,k]
        d = wmma_bf16f32(a, b, d);
    }
    int c = lane & 15, h = lane >> 4;
    float bval = bias[col0 + c];
    if (mat < 2) {
        bf16* dst = (mat == 0) ? qb : kb;
        for (int r = 0; r < 8; ++r)
            dst[(size_t)(row0 + h * 8 + r) * CHN + col0 + c] = (bf16)(d[r] + bval);
    } else {
        for (int r = 0; r < 8; ++r)         // V transposed: vT[channel][token]
            vT[(size_t)(col0 + c) * LSEQ + row0 + h * 8 + r] = (bf16)(d[r] + bval);
    }
}

// ---------------- Kernel 2: pass 1 — streaming log-sum-exp over Q K^T / L ----------------
// One workgroup per 16-row strip; each wave sweeps 64 column tiles reusing its Q fragments.
// Per-lane running (m, S) across tiles — NO cross-lane traffic in the hot loop; single
// shfl_xor merge tree per wave at the end.

__global__ void k_pass1(const bf16* __restrict__ qb, const bf16* __restrict__ kb,
                        float* partials) {
    int lane = threadIdx.x & 31, wave = threadIdx.x >> 5;
    int row0 = blockIdx.x * 16;

    v16bf aq[8];
    for (int k = 0; k < 8; ++k) aq[k] = load_a_frag(qb, CHN, row0, k * 32, lane);

    float mrun = -__builtin_inff(), srun = 0.0f;
    for (int ct = wave; ct < 512; ct += 8) {
        int col0 = ct * 16;
        if (ct + 8 < 512)   // warm GL2 for the next K-row block (global_prefetch_b8)
            __builtin_prefetch(kb + (size_t)(col0 + 128 + (lane & 15)) * CHN, 0, 0);
        v8f s = {};
        for (int k = 0; k < 8; ++k) {
            v16bf b = load_b_frag(kb, CHN, col0, k * 32, lane);  // B[k,j] = K[j,k]
            s = wmma_bf16f32(aq[k], b, s);
        }
        float z[8], mt = -__builtin_inff();
        for (int r = 0; r < 8; ++r) { z[r] = s[r] * INV_L; mt = fmaxf(mt, z[r]); }
        float nm  = fmaxf(mrun, mt);
        float acc = srun * __expf(mrun - nm);
        for (int r = 0; r < 8; ++r) acc += __expf(z[r] - nm);
        srun = acc;
        mrun = nm;
    }
    // merge (m,S) across the 32 lanes once
    for (int off = 16; off >= 1; off >>= 1) {
        float mo = __shfl_xor(mrun, off, 32);
        float so = __shfl_xor(srun, off, 32);
        float nm = fmaxf(mrun, mo);
        srun = srun * __expf(mrun - nm) + so * __expf(mo - nm);
        mrun = nm;
    }
    if (lane == 0) {
        int w = blockIdx.x * 8 + wave;      // 512*8 = 4096 partials
        partials[2 * w]     = mrun;
        partials[2 * w + 1] = srun;
    }
}

// ---------------- Kernel 3: reduce 4096 (m,S) partials to global (m,S) ----------------

__global__ void k_reduce(const float* __restrict__ partials, float* scal) {
    __shared__ float sm[256], ss[256];
    int tid = threadIdx.x;
    float m = -__builtin_inff(), s = 0.0f;
    for (int i = tid; i < 4096; i += 256) {
        float mi = partials[2 * i], si = partials[2 * i + 1];
        float nm = fmaxf(m, mi);
        s = s * __expf(m - nm) + si * __expf(mi - nm);
        m = nm;
    }
    sm[tid] = m; ss[tid] = s;
    __syncthreads();
    for (int off = 128; off >= 1; off >>= 1) {
        if (tid < off) {
            float mo = sm[tid + off], so = ss[tid + off];
            float nm = fmaxf(sm[tid], mo);
            ss[tid] = ss[tid] * __expf(sm[tid] - nm) + so * __expf(mo - nm);
            sm[tid] = nm;
        }
        __syncthreads();
    }
    if (tid == 0) { scal[0] = sm[0]; scal[1] = ss[0]; }
}

// ---------------- Kernel 4: pass 2 — out = exp(QK^T/L - m) @ V / S ----------------
// One workgroup per 16 output rows. Each chunk: 8 waves build a 16x128 bf16 P tile in
// LDS (score WMMAs + exp), then every wave drains it into its 2 output-channel tiles.

__global__ void k_pass2(const bf16* __restrict__ qb, const bf16* __restrict__ kb,
                        const bf16* __restrict__ vT, const float* __restrict__ scal,
                        float* __restrict__ out) {
    __shared__ __align__(16) unsigned short ldsP_s[16 * 128];
    bf16* ldsP = (bf16*)ldsP_s;

    int lane = threadIdx.x & 31, wave = threadIdx.x >> 5;
    int row0 = blockIdx.x * 16;
    int c = lane & 15, h = lane >> 4;

    float gm   = scal[0];
    float invS = 1.0f / scal[1];

    v16bf aq[8];
    for (int k = 0; k < 8; ++k) aq[k] = load_a_frag(qb, CHN, row0, k * 32, lane);

    v8f acc0 = {}, acc1 = {};
    int n0 = wave * 32, n1 = wave * 32 + 16;    // this wave's output channel tiles

    for (int jc = 0; jc < 64; ++jc) {           // 64 chunks of 128 keys
        int col0 = jc * 128 + wave * 16;
        v8f s = {};
        for (int k = 0; k < 8; ++k) {
            v16bf b = load_b_frag(kb, CHN, col0, k * 32, lane);
            s = wmma_bf16f32(aq[k], b, s);
        }
        // P = exp(s/L - m) into LDS, C-layout -> row-major [16 x 128] bf16
        for (int r = 0; r < 8; ++r) {
            float p = __expf(s[r] * INV_L - gm);
            ldsP[(h * 8 + r) * 128 + wave * 16 + c] = (bf16)p;
        }
        __syncthreads();
        for (int ks = 0; ks < 4; ++ks) {        // K-chunk 128 = 4 x 32
            const bf16* pa = ldsP + (size_t)(lane & 15) * 128 + ks * 32 + h * 8;
            v8bf lo = *(const v8bf*)pa;
            v8bf hi = *(const v8bf*)(pa + 16);
            v16bf a = __builtin_shufflevector(lo, hi,
                        0,1,2,3,4,5,6,7,8,9,10,11,12,13,14,15);
            int k0 = jc * 128 + ks * 32;
            v16bf b0 = load_b_frag(vT, (size_t)LSEQ, n0, k0, lane);  // B[j,c] = vT[c][j]
            acc0 = wmma_bf16f32(a, b0, acc0);
            v16bf b1 = load_b_frag(vT, (size_t)LSEQ, n1, k0, lane);
            acc1 = wmma_bf16f32(a, b1, acc1);
        }
        __syncthreads();
    }
    for (int r = 0; r < 8; ++r) {
        int row = row0 + h * 8 + r;
        out[(size_t)row * CHN + n0 + c] = acc0[r] * invS;
        out[(size_t)row * CHN + n1 + c] = acc1[r] * invS;
    }
}

// ---------------- launch ----------------

extern "C" void kernel_launch(void* const* d_in, const int* in_sizes, int n_in,
                              void* d_out, int out_size, void* d_ws, size_t ws_size,
                              hipStream_t stream) {
    (void)in_sizes; (void)n_in; (void)out_size; (void)ws_size;
    const float* x  = (const float*)d_in[0];
    const float* Wq = (const float*)d_in[1];
    const float* bq = (const float*)d_in[2];
    const float* Wk = (const float*)d_in[3];
    const float* bk = (const float*)d_in[4];
    const float* Wv = (const float*)d_in[5];
    const float* bv = (const float*)d_in[6];
    float* out = (float*)d_out;

    char* ws = (char*)d_ws;
    bf16*  xb   = (bf16*)(ws + OFF_XB);
    bf16*  wqb  = (bf16*)(ws + OFF_WQB);
    bf16*  wkb  = (bf16*)(ws + OFF_WKB);
    bf16*  wvb  = (bf16*)(ws + OFF_WVB);
    bf16*  qb   = (bf16*)(ws + OFF_QB);
    bf16*  kb   = (bf16*)(ws + OFF_KB);
    bf16*  vT   = (bf16*)(ws + OFF_VT);
    float* part = (float*)(ws + OFF_PART);
    float* scal = (float*)(ws + OFF_SCAL);

    k_prep  <<<2048, 256, 0, stream>>>(x, Wq, Wk, Wv, xb, wqb, wkb, wvb);
    k_proj  <<<dim3(1024, 3), 256, 0, stream>>>(xb, wqb, wkb, wvb, bq, bk, bv, qb, kb, vT);
    k_pass1 <<<512, 256, 0, stream>>>(qb, kb, part);
    k_reduce<<<1, 256, 0, stream>>>(part, scal);
    k_pass2 <<<512, 256, 0, stream>>>(qb, kb, vT, scal, out);
}
